// ChromatinTransformer_82231443849543
// MI455X (gfx1250) — matrix operand order repaired and checked by reading, not compiled, not run on hardware
//
#include <hip/hip_runtime.h>
#include <hip/hip_bf16.h>
#include <stdint.h>
#include <stddef.h>

// ---------------------------------------------------------------------------
// Types for CDNA5 WMMA (wave32): v_wmma_f32_16x16x32_bf16
// ---------------------------------------------------------------------------
typedef __bf16 bf16_t;
typedef bf16_t v8bf  __attribute__((ext_vector_type(8)));
typedef bf16_t v16bf __attribute__((ext_vector_type(16)));
typedef float  v8f   __attribute__((ext_vector_type(8)));

static __device__ __forceinline__ bf16_t f2bf(float f) {
  union { float f; unsigned u; } a; a.f = f;
  unsigned r = a.u + 0x7FFFu + ((a.u >> 16) & 1u);      // round to nearest even
  union { unsigned short s; bf16_t b; } o;
  o.s = (unsigned short)(r >> 16);
  return o.b;
}

// Order-preserving float -> uint mapping for atomic max (softmax stability)
static __device__ __forceinline__ unsigned fmap(float f) {
  unsigned i = __float_as_uint(f);
  return (i & 0x80000000u) ? ~i : (i | 0x80000000u);
}
static __device__ __forceinline__ float funmap(unsigned u) {
  unsigned i = (u & 0x80000000u) ? (u & 0x7FFFFFFFu) : ~u;
  return __uint_as_float(i);
}

// ---------------------------------------------------------------------------
// Utility fill (also used to zero float buffers bit-wise)
// ---------------------------------------------------------------------------
__global__ void k_fill_u32(unsigned* __restrict__ p, unsigned v, long n) {
  long i = (long)blockIdx.x * blockDim.x + threadIdx.x;
  if (i < n) p[i] = v;
}

// ---------------------------------------------------------------------------
// f32 activations -> zero-padded bf16 A matrix [Mp x Kp]
// ---------------------------------------------------------------------------
__global__ void k_cvt_bf16(const float* __restrict__ X, bf16_t* __restrict__ A,
                           int Nreal, int fin, int Kp, int Mp) {
  long idx = (long)blockIdx.x * blockDim.x + threadIdx.x;
  long total = (long)Mp * Kp;
  if (idx >= total) return;
  int r = (int)(idx / Kp), k = (int)(idx % Kp);
  float v = (r < Nreal && k < fin) ? X[(size_t)r * fin + k] : 0.0f;
  A[idx] = f2bf(v);
}

// ---------------------------------------------------------------------------
// Pack Wq|Wk|Wv|Ws (each [fin x cols], f32) into transposed bf16 [Ntot x Kp],
// gather biases into BIAS[Ntot].
// ---------------------------------------------------------------------------
__global__ void k_pack_w(const float* __restrict__ Wq, const float* __restrict__ Wk,
                         const float* __restrict__ Wv, const float* __restrict__ Ws,
                         const float* __restrict__ bq, const float* __restrict__ bk,
                         const float* __restrict__ bv, const float* __restrict__ bs,
                         bf16_t* __restrict__ WT, float* __restrict__ BIAS,
                         int fin, int Kp, int hc, int sc) {
  int Ntot = 3 * hc + sc;
  int idx = blockIdx.x * blockDim.x + threadIdx.x;
  if (idx >= Ntot * Kp) return;
  int n = idx / Kp, k = idx % Kp;
  const float *W, *b; int cols, nloc;
  if      (n <     hc) { W = Wq; b = bq; cols = hc; nloc = n;          }
  else if (n < 2 * hc) { W = Wk; b = bk; cols = hc; nloc = n - hc;     }
  else if (n < 3 * hc) { W = Wv; b = bv; cols = hc; nloc = n - 2 * hc; }
  else                 { W = Ws; b = bs; cols = sc; nloc = n - 3 * hc; }
  float w = (k < fin) ? W[(size_t)k * cols + nloc] : 0.0f;
  WT[(size_t)n * Kp + k] = f2bf(w);
  if (k == 0) BIAS[n] = b[nloc];
}

// ---------------------------------------------------------------------------
// WMMA GEMM: OUT[M x Ntot] (f32) = A[Mp x Kp] (bf16) @ WT^T + BIAS
// Block = 4 waves = 64 rows x 128 cols. The 128-col weight panel (contiguous
// chunk of WT) is staged into LDS once per block via the CDNA5 async-to-LDS
// path (global_load_async_to_lds_b128 + s_wait_asynccnt), then the K-loop
// reads B fragments from LDS (ds_load_b128, padded stride for bank spread)
// while A fragments are software-pipelined from global.
// ---------------------------------------------------------------------------
__global__ void __launch_bounds__(128)
k_gemm_wmma(const bf16_t* __restrict__ A, const bf16_t* __restrict__ WT,
            const float* __restrict__ BIAS, float* __restrict__ OUT,
            int Mreal, int Kp, int Ntot, int kvsh /* log2(Kp/8) */) {
  __shared__ bf16_t Bs[128 * (128 + 8)];   // [col][k], row stride Kp+8 elems

  const int tid   = threadIdx.x;
  const int lane  = tid & 31;
  const int wave  = tid >> 5;
  const int tileM = blockIdx.x * 64 + wave * 16;
  const int colB  = blockIdx.y * 128;
  const int r  = lane & 15;
  const int kh = lane >> 4;
  const int bstride = Kp + 8;

  // ---- stage B panel into LDS (async, per-lane 16B vectors) ----
  {
    const int kvecs = 1 << kvsh;          // 16B vectors per column row
    const int nvec  = 128 << kvsh;        // total 16B vectors in panel
    unsigned ldsbase = (unsigned)(size_t)(&Bs[0]);
    for (int i = tid; i < nvec; i += 128) {
      int col = i >> kvsh;
      int kv  = i & (kvecs - 1);
      unsigned loff = ldsbase + (unsigned)((col * bstride + kv * 8) * 2);
      const bf16_t* g = WT + (size_t)(colB + col) * Kp + kv * 8;
      asm volatile("global_load_async_to_lds_b128 %0, %1, off"
                   :: "v"(loff), "v"(g) : "memory");
    }
    asm volatile("s_wait_asynccnt 0x0" ::: "memory");
    __syncthreads();
  }

  v8f acc[8] = {};

  const bf16_t* arow = A + (size_t)(tileM + r) * Kp;
  v8bf a0 = *(const v8bf*)(arow + kh * 8);
  v8bf a1 = *(const v8bf*)(arow + 16 + kh * 8);

  for (int kk = 0; kk < Kp; kk += 32) {
    v16bf af;
#pragma unroll
    for (int i = 0; i < 8; ++i) { af[i] = a0[i]; af[8 + i] = a1[i]; }
    if (kk + 32 < Kp) {                    // pipeline next A fragment
      a0 = *(const v8bf*)(arow + kk + 32 + kh * 8);
      a1 = *(const v8bf*)(arow + kk + 48 + kh * 8);
    }
    // preload all 8 B fragments of this K-step from LDS
    v16bf bfr[8];
#pragma unroll
    for (int g = 0; g < 8; ++g) {
      const bf16_t* bc = &Bs[(g * 16 + r) * bstride + kk + kh * 8];
      v8bf b0 = *(const v8bf*)(bc);
      v8bf b1 = *(const v8bf*)(bc + 16);
#pragma unroll
      for (int i = 0; i < 8; ++i) { bfr[g][i] = b0[i]; bfr[g][8 + i] = b1[i]; }
    }
#pragma unroll
    for (int g = 0; g < 8; ++g) {
      acc[g] = __builtin_amdgcn_wmma_f32_16x16x32_bf16(
          /*neg_a=*/false, af, /*neg_b=*/false, bfr[g],
          /*c_mod=*/(short)0, acc[g], /*reuse_a=*/false, /*reuse_b=*/false);
    }
  }

  // C/D layout: VGPR i -> row = i + kh*8, col = colB + g*16 + (lane&15)
#pragma unroll
  for (int g = 0; g < 8; ++g) {
    int col = colB + g * 16 + r;
    float bias = BIAS[col];
#pragma unroll
    for (int i = 0; i < 8; ++i) {
      int row = tileM + i + kh * 8;
      if (row < Mreal) OUT[(size_t)row * Ntot + col] = acc[g][i] + bias;
    }
  }
}

// ---------------------------------------------------------------------------
// Edge phase. PROJ row layout: [ q(hc) | k(hc) | v(hc) | skip(sc) ], stride rs.
// Templated on H (heads) and C (channels) so /H, %H fold to shifts and the
// C-loops fully unroll into float4 streams.
// ---------------------------------------------------------------------------
template <int H, int C>
__global__ void k_edge_alpha(const int* __restrict__ src, const int* __restrict__ dst,
                             const float* __restrict__ ea, const float* __restrict__ We,
                             const float* __restrict__ PROJ, int rs,
                             float* __restrict__ alpha, unsigned* __restrict__ mmax,
                             int E) {
  const int hc = H * C;
  int idx = blockIdx.x * blockDim.x + threadIdx.x;
  if (idx >= E * H) return;
  int e = idx / H, hh = idx % H;
  int s = src[e], d = dst[e];
  float eav = ea[e];
  const float* q  = PROJ + (size_t)d * rs + hh * C;
  const float* kv = PROJ + (size_t)s * rs + hc + hh * C;
  const float* we = We + hh * C;
  float acc = 0.0f;
#pragma unroll
  for (int j = 0; j < C; j += 4) {
    float4 qv = *(const float4*)(q + j);
    float4 kk = *(const float4*)(kv + j);
    float4 w  = *(const float4*)(we + j);
    acc += qv.x * (kk.x + eav * w.x) + qv.y * (kk.y + eav * w.y) +
           qv.z * (kk.z + eav * w.z) + qv.w * (kk.w + eav * w.w);
  }
  acc *= rsqrtf((float)C);
  alpha[idx] = acc;
  atomicMax(mmax + (size_t)d * H + hh, fmap(acc));
}

template <int H>
__global__ void k_edge_exp(float* __restrict__ alpha, const unsigned* __restrict__ mmax,
                           float* __restrict__ den, const int* __restrict__ dst, int E) {
  int idx = blockIdx.x * blockDim.x + threadIdx.x;
  if (idx >= E * H) return;
  int e = idx / H, hh = idx % H;
  int d = dst[e];
  float m = funmap(mmax[(size_t)d * H + hh]);
  float ex = __expf(alpha[idx] - m);
  alpha[idx] = ex;
  atomicAdd(den + (size_t)d * H + hh, ex);
}

template <int H, int C>
__global__ void k_edge_scatter(const float* __restrict__ exv, const float* __restrict__ den,
                               const int* __restrict__ src, const int* __restrict__ dst,
                               const float* __restrict__ ea, const float* __restrict__ We,
                               const float* __restrict__ PROJ, int rs,
                               float* __restrict__ AGG, int E) {
  const int hc = H * C;
  int idx = blockIdx.x * blockDim.x + threadIdx.x;
  if (idx >= E * H) return;
  int e = idx / H, hh = idx % H;
  int s = src[e], d = dst[e];
  float a = exv[idx] / (den[(size_t)d * H + hh] + 1e-16f);
  float eav = ea[e];
  const float* v  = PROJ + (size_t)s * rs + 2 * hc + hh * C;
  const float* we = We + hh * C;
  float* o = AGG + (size_t)d * hc + hh * C;
#pragma unroll
  for (int j = 0; j < C; j += 4) {
    float4 vv = *(const float4*)(v + j);
    float4 w  = *(const float4*)(we + j);
    atomicAdd(o + j + 0, (vv.x + eav * w.x) * a);
    atomicAdd(o + j + 1, (vv.y + eav * w.y) * a);
    atomicAdd(o + j + 2, (vv.z + eav * w.z) * a);
    atomicAdd(o + j + 3, (vv.w + eav * w.w) * a);
  }
}

// combine: concat (or head-mean) + skip -> XOUT [N x dout]
__global__ void k_combine(const float* __restrict__ AGG, const float* __restrict__ PROJ,
                          float* __restrict__ XOUT, int N, int h, int c, int concat, int rs) {
  int hc = h * c;
  int dout = concat ? hc : c;
  long idx = (long)blockIdx.x * blockDim.x + threadIdx.x;
  if (idx >= (long)N * dout) return;
  int n = (int)(idx / dout), dd = (int)(idx % dout);
  float val;
  if (concat) {
    val = AGG[(size_t)n * hc + dd];
  } else {
    val = 0.0f;
    for (int hh = 0; hh < h; ++hh) val += AGG[(size_t)n * hc + hh * c + dd];
    val *= (1.0f / h);
  }
  val += PROJ[(size_t)n * rs + 3 * hc + dd];
  XOUT[idx] = val;
}

// BatchNorm stats: per-column sum / sumsq, hierarchical reduce + one atomic/block
__global__ void k_bn_stats(const float* __restrict__ X, float* __restrict__ sum,
                           float* __restrict__ sumsq, int N, int dout) {
  __shared__ float s1[256];
  __shared__ float s2[256];
  const int col = blockIdx.x;
  const int CH = 8192;
  int r0 = blockIdx.y * CH;
  int r1 = r0 + CH; if (r1 > N) r1 = N;
  float a = 0.0f, b = 0.0f;
  for (int r = r0 + threadIdx.x; r < r1; r += blockDim.x) {
    float v = X[(size_t)r * dout + col];
    a += v; b += v * v;
  }
  s1[threadIdx.x] = a; s2[threadIdx.x] = b;
  __syncthreads();
  for (int s = 128; s > 0; s >>= 1) {
    if ((int)threadIdx.x < s) {
      s1[threadIdx.x] += s1[threadIdx.x + s];
      s2[threadIdx.x] += s2[threadIdx.x + s];
    }
    __syncthreads();
  }
  if (threadIdx.x == 0) { atomicAdd(&sum[col], s1[0]); atomicAdd(&sumsq[col], s2[0]); }
}

__global__ void k_bn_apply_elu(float* __restrict__ X, const float* __restrict__ sum,
                               const float* __restrict__ sumsq,
                               const float* __restrict__ gamma, const float* __restrict__ beta,
                               int N, int dout) {
  long idx = (long)blockIdx.x * blockDim.x + threadIdx.x;
  if (idx >= (long)N * dout) return;
  int d = (int)(idx % dout);
  float inv = 1.0f / (float)N;
  float mu = sum[d] * inv;
  float var = sumsq[d] * inv - mu * mu;
  float y = (X[idx] - mu) * rsqrtf(var + 1e-5f) * gamma[d] + beta[d];
  X[idx] = (y > 0.0f) ? y : (__expf(y) - 1.0f);
}

// classifier head: [Nx32] -> relu(@W1[32x16]+b1) -> @W2[16x2]+b2
__global__ void k_cls(const float* __restrict__ X, const float* __restrict__ W1,
                      const float* __restrict__ b1, const float* __restrict__ W2,
                      const float* __restrict__ b2, float* __restrict__ OUT, int N) {
  int n = blockIdx.x * blockDim.x + threadIdx.x;
  if (n >= N) return;
  const float* x = X + (size_t)n * 32;
  float hbuf[16];
#pragma unroll
  for (int j = 0; j < 16; ++j) {
    float s = b1[j];
    for (int k = 0; k < 32; ++k) s += x[k] * W1[k * 16 + j];
    hbuf[j] = s > 0.0f ? s : 0.0f;
  }
#pragma unroll
  for (int o = 0; o < 2; ++o) {
    float s = b2[o];
    for (int j = 0; j < 16; ++j) s += hbuf[j] * W2[j * 2 + o];
    OUT[(size_t)n * 2 + o] = s;
  }
}

// ---------------------------------------------------------------------------
// Host orchestration
// ---------------------------------------------------------------------------
static inline unsigned cdiv(long n, int b) { return (unsigned)((n + b - 1) / b); }

extern "C" void kernel_launch(void* const* d_in, const int* in_sizes, int n_in,
                              void* d_out, int out_size, void* d_ws, size_t ws_size,
                              hipStream_t stream) {
  (void)n_in; (void)out_size; (void)ws_size;

  const int N = in_sizes[0] / 16;      // 50000
  const int E = in_sizes[1] / 2;       // 800000
  const int Mp = ((N + 63) / 64) * 64; // 50048 (A padded so every wave's WMMA rows exist)

  const float* x_in = (const float*)d_in[0];
  const int*   src  = (const int*)d_in[1];
  const int*   dst  = src + E;
  const float* ea   = (const float*)d_in[2];

  // ---- carve workspace ----
  size_t off = 0;
  auto carve = [&](size_t bytes) -> void* {
    void* p = (char*)d_ws + off;
    off += (bytes + 255) & ~(size_t)255;
    return p;
  };
  bf16_t* A_BF  = (bf16_t*)carve((size_t)Mp * 128 * sizeof(bf16_t));
  float*  PROJ  = (float*) carve((size_t)N * 512 * sizeof(float));
  float*  XB0   = (float*) carve((size_t)N * 128 * sizeof(float));
  float*  XB1   = (float*) carve((size_t)N * 128 * sizeof(float));
  float*  AGG   = (float*) carve((size_t)N * 128 * sizeof(float));
  float*  ALPHA = (float*) carve((size_t)E * 4 * sizeof(float));
  unsigned* MMAX = (unsigned*)carve((size_t)N * 4 * sizeof(unsigned));
  float*  DEN   = (float*) carve((size_t)N * 4 * sizeof(float));
  bf16_t* WT    = (bf16_t*)carve((size_t)512 * 128 * sizeof(bf16_t));
  float*  BIAS  = (float*) carve(512 * sizeof(float));
  float*  SUM   = (float*) carve(128 * sizeof(float));
  float*  SQ    = (float*) carve(128 * sizeof(float));

  struct Cfg { int fin, h, c, concat, base; };
  const Cfg cfgs[3] = { {16, 4, 32, 1, 3}, {128, 4, 32, 1, 12}, {128, 1, 32, 0, 21} };

  const float* xcur = x_in;
  float* xbuf[2] = { XB0, XB1 };
  float* xout = XB0;

  for (int L = 0; L < 3; ++L) {
    const Cfg cg = cfgs[L];
    const int hc = cg.h * cg.c;
    const int sc = cg.concat ? hc : cg.c;
    const int Ntot = 3 * hc + sc;                 // 512, 512, 128
    const int Kp = (cg.fin <= 32) ? 32 : ((cg.fin + 31) / 32) * 32;
    const int kvsh = (Kp == 32) ? 2 : 4;          // log2(Kp/8)
    const int b = cg.base;
    const float* Wq = (const float*)d_in[b + 0];
    const float* bq = (const float*)d_in[b + 1];
    const float* Wk = (const float*)d_in[b + 2];
    const float* bk = (const float*)d_in[b + 3];
    const float* Wv = (const float*)d_in[b + 4];
    const float* bv = (const float*)d_in[b + 5];
    const float* We = (const float*)d_in[b + 6];
    const float* Ws = (const float*)d_in[b + 7];
    const float* bs = (const float*)d_in[b + 8];
    const float* gamma = (const float*)d_in[30 + 2 * L];
    const float* beta  = (const float*)d_in[31 + 2 * L];

    // 1) activations -> padded bf16 A
    k_cvt_bf16<<<cdiv((long)Mp * Kp, 256), 256, 0, stream>>>(xcur, A_BF, N, cg.fin, Kp, Mp);
    // 2) pack fused transposed bf16 weights + bias
    k_pack_w<<<cdiv((long)Ntot * Kp, 256), 256, 0, stream>>>(
        Wq, Wk, Wv, Ws, bq, bk, bv, bs, WT, BIAS, cg.fin, Kp, hc, sc);
    // 3) fused Q|K|V|skip projection on WMMA (B panel staged via async-to-LDS)
    dim3 gg(cdiv(Mp, 64), Ntot / 128);
    k_gemm_wmma<<<gg, 128, 0, stream>>>(A_BF, WT, BIAS, PROJ, N, Kp, Ntot, kvsh);

    // 4) softmax scratch init
    k_fill_u32<<<cdiv((long)N * cg.h, 256), 256, 0, stream>>>(MMAX, 0u, (long)N * cg.h);
    k_fill_u32<<<cdiv((long)N * cg.h, 256), 256, 0, stream>>>((unsigned*)DEN, 0u, (long)N * cg.h);
    k_fill_u32<<<cdiv((long)N * hc, 256), 256, 0, stream>>>((unsigned*)AGG, 0u, (long)N * hc);

    // 5-7) edge attention: alpha+max, exp+sum, weighted scatter
    unsigned eg = cdiv((long)E * cg.h, 256);
    if (cg.h == 4) {
      k_edge_alpha<4, 32><<<eg, 256, 0, stream>>>(src, dst, ea, We, PROJ, Ntot, ALPHA, MMAX, E);
      k_edge_exp<4><<<eg, 256, 0, stream>>>(ALPHA, MMAX, DEN, dst, E);
      k_edge_scatter<4, 32><<<eg, 256, 0, stream>>>(ALPHA, DEN, src, dst, ea, We, PROJ, Ntot, AGG, E);
    } else {
      k_edge_alpha<1, 32><<<eg, 256, 0, stream>>>(src, dst, ea, We, PROJ, Ntot, ALPHA, MMAX, E);
      k_edge_exp<1><<<eg, 256, 0, stream>>>(ALPHA, MMAX, DEN, dst, E);
      k_edge_scatter<1, 32><<<eg, 256, 0, stream>>>(ALPHA, DEN, src, dst, ea, We, PROJ, Ntot, AGG, E);
    }

    // 8) combine + skip
    const int dout = cg.concat ? hc : cg.c;
    xout = xbuf[L & 1];
    k_combine<<<cdiv((long)N * dout, 256), 256, 0, stream>>>(AGG, PROJ, xout, N, cg.h, cg.c, cg.concat, Ntot);

    // 9) BatchNorm + ELU
    k_fill_u32<<<1, 256, 0, stream>>>((unsigned*)SUM, 0u, 128);
    k_fill_u32<<<1, 256, 0, stream>>>((unsigned*)SQ, 0u, 128);
    k_bn_stats<<<dim3(dout, cdiv(N, 8192)), 256, 0, stream>>>(xout, SUM, SQ, N, dout);
    k_bn_apply_elu<<<cdiv((long)N * dout, 256), 256, 0, stream>>>(xout, SUM, SQ, gamma, beta, N, dout);

    xcur = xout;
  }

  // classifier head -> d_out [N x 2] f32
  const float* W1 = (const float*)d_in[36];
  const float* b1 = (const float*)d_in[37];
  const float* W2 = (const float*)d_in[38];
  const float* b2 = (const float*)d_in[39];
  k_cls<<<cdiv(N, 256), 256, 0, stream>>>(xcur, W1, b1, W2, b2, (float*)d_out, N);
}
